// MultiHeadAttention_80470507258396
// MI455X (gfx1250) — matrix operand order repaired and checked by reading, not compiled, not run on hardware
//
#include <hip/hip_runtime.h>
#include <hip/hip_bf16.h>

typedef __attribute__((ext_vector_type(16))) __bf16 v16bf;
typedef __attribute__((ext_vector_type(8)))  __bf16 v8bf;
typedef __attribute__((ext_vector_type(8)))  float  v8f;

static __device__ __forceinline__ unsigned pack2(__bf16 a, __bf16 b) {
    unsigned short x = __builtin_bit_cast(unsigned short, a);
    unsigned short y = __builtin_bit_cast(unsigned short, b);
    return (unsigned)x | ((unsigned)y << 16);
}

static __device__ __forceinline__ v16bf cat8(v8bf lo, v8bf hi) {
    return __builtin_shufflevector(lo, hi, 0,1,2,3,4,5,6,7,8,9,10,11,12,13,14,15);
}

// CDNA5 async global->LDS copy (ASYNCcnt-tracked), see CDNA5 ISA
static __device__ __forceinline__ void async_b128(unsigned lds_off, const void* g) {
    asm volatile("global_load_async_to_lds_b128 %0, %1, off"
                 :: "v"(lds_off), "v"(g) : "memory");
}
static __device__ __forceinline__ void wait_async0() {
    asm volatile("s_wait_asynccnt 0x0" ::: "memory");
}

#define WMMA_BF16(A,B,C) \
    __builtin_amdgcn_wmma_f32_16x16x32_bf16(false,(A),false,(B),(short)0,(C),false,false)

// ---------------------------------------------------------------------------
// GEMM: Y = (X @ W^T + bias) * oscale.  X:(M,K) fp32 or bf16, W:(N,K) fp32.
// Block = 256 threads (8 waves). Tile = 128 x 64, K-step 32. One wave -> 16x64.
// SPLIT_HEADS: write bf16 into (B=4, h=16, S=2048, dk=64); else fp32 (M,N).
// oscale lets the 1/sqrt(dk) attention scale be folded into the Q projection
// (power-of-two => exact in bf16).
// ---------------------------------------------------------------------------
template<bool IN_BF16, bool SPLIT_HEADS>
__global__ void __launch_bounds__(256)
mha_gemm_wmma(const void* __restrict__ Xv, const float* __restrict__ W,
              const float* __restrict__ bias, void* __restrict__ Yv,
              int M, int N, int K, float oscale)
{
    __shared__ __bf16 Xs[128 * 32];   // A tile (row-major, K=32)
    __shared__ __bf16 Ws[32 * 64];    // B tile: Ws[k][n] = W[n0+n][kk+k]

    const int tid  = threadIdx.x;
    const int wave = tid >> 5;
    const int lane = tid & 31;
    const int m0 = blockIdx.x * 128;
    const int n0 = blockIdx.y * 64;

    const unsigned Xs_base = (unsigned)(uintptr_t)&Xs[0];

    v8f acc[4] = {};

    for (int kk = 0; kk < K; kk += 32) {
        __syncthreads();
        {   // stage X tile (128 x 32)
            const int r  = tid >> 1;          // 0..127
            const int cb = (tid & 1) * 16;    // 0 or 16
            if (IN_BF16) {
                // straight bf16 copy: use CDNA5 async global->LDS datapath
                const __bf16* src = (const __bf16*)Xv + (size_t)(m0 + r) * K + kk + cb;
                const unsigned loff = Xs_base + (unsigned)(r * 64 + cb * 2);
                async_b128(loff,      src);
                async_b128(loff + 16, src + 8);
            } else {
                // fp32 -> bf16 convert with packed dword LDS stores
                const float* src = (const float*)Xv + (size_t)(m0 + r) * K + kk + cb;
                float4 f0 = *(const float4*)(src);
                float4 f1 = *(const float4*)(src + 4);
                float4 f2 = *(const float4*)(src + 8);
                float4 f3 = *(const float4*)(src + 12);
                unsigned* dst = (unsigned*)&Xs[r * 32 + cb];
                dst[0] = pack2((__bf16)f0.x, (__bf16)f0.y);
                dst[1] = pack2((__bf16)f0.z, (__bf16)f0.w);
                dst[2] = pack2((__bf16)f1.x, (__bf16)f1.y);
                dst[3] = pack2((__bf16)f1.z, (__bf16)f1.w);
                dst[4] = pack2((__bf16)f2.x, (__bf16)f2.y);
                dst[5] = pack2((__bf16)f2.z, (__bf16)f2.w);
                dst[6] = pack2((__bf16)f3.x, (__bf16)f3.y);
                dst[7] = pack2((__bf16)f3.z, (__bf16)f3.w);
            }
        }
        {   // stage W tile transposed; adjacent n-pair -> packed b32 LDS stores
            const int n2 = (tid >> 3) * 2;    // 0,2,..,62
            const int kb = (tid & 7) * 4;     // 0,4,..,28
            const float* s0 = W + (size_t)(n0 + n2) * K + kk + kb;
            const float* s1 = s0 + K;
            float4 r0 = *(const float4*)s0;
            float4 r1 = *(const float4*)s1;
            if (kk + 32 < K) __builtin_prefetch(s0 + 32, 0, 0);
            unsigned p0 = pack2((__bf16)r0.x, (__bf16)r1.x);
            unsigned p1 = pack2((__bf16)r0.y, (__bf16)r1.y);
            unsigned p2 = pack2((__bf16)r0.z, (__bf16)r1.z);
            unsigned p3 = pack2((__bf16)r0.w, (__bf16)r1.w);
            *(unsigned*)&Ws[(kb + 0) * 64 + n2] = p0;
            *(unsigned*)&Ws[(kb + 1) * 64 + n2] = p1;
            *(unsigned*)&Ws[(kb + 2) * 64 + n2] = p2;
            *(unsigned*)&Ws[(kb + 3) * 64 + n2] = p3;
        }
        if (IN_BF16) wait_async0();
        __syncthreads();

        // A fragment: lane = row (M), chunks at (lane/16)*8 and +16 (ISA A-layout)
        const __bf16* ap = &Xs[(wave * 16 + (lane & 15)) * 32 + ((lane >> 4) * 8)];
        v16bf afrag = cat8(*(const v8bf*)ap, *(const v8bf*)(ap + 16));
        #pragma unroll
        for (int nt = 0; nt < 4; ++nt) {
            // B fragment: lane = N column, K chunks at (lane/16)*8 and +16 (ISA B-layout)
            v16bf bfrag;
            #pragma unroll
            for (int s = 0; s < 16; ++s)
                bfrag[s] = Ws[((lane >> 4) * 8 + (s & 7) + (s >> 3) * 16) * 64 + nt * 16 + (lane & 15)];
            acc[nt] = WMMA_BF16(afrag, bfrag, acc[nt]);
        }
    }

    // Epilogue: C/D layout -> VGPR j holds row j (lanes 0-15) / j+8 (lanes 16-31)
    #pragma unroll
    for (int nt = 0; nt < 4; ++nt) {
        const int n = n0 + nt * 16 + (lane & 15);
        const float bval = bias[n];
        #pragma unroll
        for (int j = 0; j < 8; ++j) {
            const int m = m0 + wave * 16 + ((lane >> 4) ? (j + 8) : j);
            const float val = (acc[nt][j] + bval) * oscale;
            if (SPLIT_HEADS) {
                const int bb = m >> 11, s = m & 2047;
                const int h  = n >> 6,  d = n & 63;
                ((__bf16*)Yv)[((size_t)(bb * 16 + h) * 2048 + s) * 64 + d] = (__bf16)val;
            } else {
                ((float*)Yv)[(size_t)m * N + n] = val;
            }
        }
    }
}

// ---------------------------------------------------------------------------
// Flash attention: per (b,h), 128 query rows per block (16 per wave),
// loop over key tiles of 64.  qh (pre-scaled by 1/sqrt(dk)) /kh/vh:
// (B*h, 2048, 64) bf16.  ctx out: (B, 2048, 1024) bf16.
// ---------------------------------------------------------------------------
__global__ void __launch_bounds__(256)
mha_attn_wmma(const __bf16* __restrict__ qh, const __bf16* __restrict__ kh,
              const __bf16* __restrict__ vh, __bf16* __restrict__ ctx)
{
    __shared__ __bf16 kT[64 * 64];        // kT[d][n] = K[k0+n][d]
    __shared__ __bf16 vS[64 * 64];        // V tile (row-major), filled by async copy
    __shared__ __bf16 Ps[8 * 16 * 64];    // per-wave P staging (D-layout -> A-layout)

    const int tid  = threadIdx.x;
    const int wave = tid >> 5;
    const int lane = tid & 31;
    const int bh   = blockIdx.y;                       // 0..63 = b*16+h
    const size_t base = (size_t)bh * 2048 * 64;
    const int qrow0 = blockIdx.x * 128 + wave * 16;

    const int ar = lane & 15;          // A-layout row
    const int ac = (lane >> 4) * 8;    // A-layout K-chunk base

    // Q fragments for the two K-steps (dk = 64 = 2 x 32)
    v16bf qf[2];
    #pragma unroll
    for (int ks = 0; ks < 2; ++ks) {
        const __bf16* p = qh + base + (size_t)(qrow0 + ar) * 64 + ks * 32 + ac;
        qf[ks] = cat8(*(const v8bf*)p, *(const v8bf*)(p + 16));
    }

    float mi[8], li[8];
    #pragma unroll
    for (int j = 0; j < 8; ++j) { mi[j] = -3.0e38f; li[j] = 0.0f; }
    v8f oacc[4] = {};

    const unsigned vS_base = (unsigned)(uintptr_t)&vS[0];

    for (int kt = 0; kt < 32; ++kt) {
        const int k0 = kt * 64;
        __syncthreads();

        // async-stage V tile (64x64 bf16 = 8KB): 2 x b128 per thread
        {
            const unsigned boff = (unsigned)tid * 16u;
            const __bf16* g0 = vh + base + (size_t)k0 * 64 + boff / 2;
            async_b128(vS_base + boff,         g0);
            async_b128(vS_base + boff + 4096u, g0 + 2048);
        }
        // stage K^T with packed b32 stores (adjacent key-pair per thread)
        {
            const int n2 = (tid >> 3) * 2;    // key pair 0,2,..,62
            const int db = (tid & 7) * 8;     // d base 0,8,..,56
            const __bf16* s0 = kh + base + (size_t)(k0 + n2) * 64 + db;
            v8bf r0 = *(const v8bf*)s0;
            v8bf r1 = *(const v8bf*)(s0 + 64);
            #pragma unroll
            for (int i = 0; i < 8; ++i)
                *(unsigned*)&kT[(db + i) * 64 + n2] = pack2(r0[i], r1[i]);
        }
        wait_async0();
        __syncthreads();

        // S = (Q K^T), scale pre-folded into Q: 4 n-tiles x 2 chained K-steps
        v8f sc[4] = {};
        #pragma unroll
        for (int nt = 0; nt < 4; ++nt) {
            #pragma unroll
            for (int ks = 0; ks < 2; ++ks) {
                v16bf bf;
                #pragma unroll
                for (int s = 0; s < 16; ++s)
                    bf[s] = kT[(ks * 32 + ac + (s & 7) + (s >> 3) * 16) * 64 + nt * 16 + ar];
                sc[nt] = WMMA_BF16(qf[ks], bf, sc[nt]);
            }
        }

        // online softmax (rows live across 16 lanes -> xor-shuffle reduce)
        #pragma unroll
        for (int j = 0; j < 8; ++j) {
            float rm = fmaxf(fmaxf(sc[0][j], sc[1][j]), fmaxf(sc[2][j], sc[3][j]));
            #pragma unroll
            for (int mk = 1; mk <= 8; mk <<= 1)
                rm = fmaxf(rm, __shfl_xor(rm, mk, 32));
            const float mnew  = fmaxf(mi[j], rm);
            const float alpha = __expf(mi[j] - mnew);
            float rs = 0.0f;
            #pragma unroll
            for (int nt = 0; nt < 4; ++nt) {
                const float p = __expf(sc[nt][j] - mnew);
                sc[nt][j] = p;
                rs += p;
            }
            #pragma unroll
            for (int mk = 1; mk <= 8; mk <<= 1)
                rs += __shfl_xor(rs, mk, 32);
            li[j] = li[j] * alpha + rs;
            mi[j] = mnew;
            #pragma unroll
            for (int dt = 0; dt < 4; ++dt) oacc[dt][j] *= alpha;
        }

        // P: D-layout regs -> per-wave LDS -> A-layout fragments
        __bf16* pw = &Ps[wave * 16 * 64];
        #pragma unroll
        for (int nt = 0; nt < 4; ++nt) {
            #pragma unroll
            for (int j = 0; j < 8; ++j) {
                const int r = (lane >> 4) ? (j + 8) : j;
                pw[r * 64 + nt * 16 + (lane & 15)] = (__bf16)sc[nt][j];
            }
        }
        #pragma unroll
        for (int ks = 0; ks < 2; ++ks) {
            const __bf16* pa = &pw[ar * 64 + ks * 32 + ac];
            v16bf af = cat8(*(const v8bf*)pa, *(const v8bf*)(pa + 16));
            #pragma unroll
            for (int dt = 0; dt < 4; ++dt) {
                v16bf bf;
                #pragma unroll
                for (int s = 0; s < 16; ++s)
                    bf[s] = vS[(ks * 32 + ac + (s & 7) + (s >> 3) * 16) * 64 + dt * 16 + ar];
                oacc[dt] = WMMA_BF16(af, bf, oacc[dt]);
            }
        }
    }

    // write ctx (B, S, 1024) bf16
    const int b = bh >> 4;
    const int h = bh & 15;
    #pragma unroll
    for (int dt = 0; dt < 4; ++dt) {
        #pragma unroll
        for (int j = 0; j < 8; ++j) {
            const int row = qrow0 + ((lane >> 4) ? (j + 8) : j);
            const float inv = 1.0f / li[j];
            ctx[((size_t)(b * 2048 + row) * 1024) + h * 64 + dt * 16 + (lane & 15)]
                = (__bf16)(oacc[dt][j] * inv);
        }
    }
}

// ---------------------------------------------------------------------------
extern "C" void kernel_launch(void* const* d_in, const int* in_sizes, int n_in,
                              void* d_out, int out_size, void* d_ws, size_t ws_size,
                              hipStream_t stream)
{
    const float* Q  = (const float*)d_in[0];
    const float* K  = (const float*)d_in[1];
    const float* V  = (const float*)d_in[2];
    const float* Wq = (const float*)d_in[3];
    const float* bq = (const float*)d_in[4];
    const float* Wk = (const float*)d_in[5];
    const float* bk = (const float*)d_in[6];
    const float* Wv = (const float*)d_in[7];
    const float* bv = (const float*)d_in[8];
    const float* Wo = (const float*)d_in[9];
    const float* bo = (const float*)d_in[10];

    const size_t nProj = (size_t)4 * 16 * 2048 * 64;   // elements per projected tensor
    __bf16* qhB  = (__bf16*)d_ws;
    __bf16* khB  = qhB + nProj;
    __bf16* vhB  = khB + nProj;
    __bf16* ctxB = vhB + nProj;

    dim3 blk(256);
    dim3 gproj(8192 / 128, 1024 / 64);
    // Q projection pre-scaled by 1/sqrt(dk) = 0.125 (exact power-of-two)
    mha_gemm_wmma<false, true ><<<gproj, blk, 0, stream>>>(Q, Wq, bq, qhB, 8192, 1024, 1024, 0.125f);
    mha_gemm_wmma<false, true ><<<gproj, blk, 0, stream>>>(K, Wk, bk, khB, 8192, 1024, 1024, 1.0f);
    mha_gemm_wmma<false, true ><<<gproj, blk, 0, stream>>>(V, Wv, bv, vhB, 8192, 1024, 1024, 1.0f);
    mha_attn_wmma<<<dim3(2048 / 128, 64), blk, 0, stream>>>(qhB, khB, vhB, ctxB);
    mha_gemm_wmma<true,  false><<<gproj, blk, 0, stream>>>(ctxB, Wo, bo, d_out, 8192, 1024, 1024, 1.0f);
}
